// SelfAttention_80367428043529
// MI455X (gfx1250) — matrix operand order repaired and checked by reading, not compiled, not run on hardware
//
#include <hip/hip_runtime.h>
#include <stdint.h>
#include <stddef.h>

#define HIDDEN 896
#define NHEADS 14
#define NKV 2
#define HD 64
#define NREP 7
#define BATCH 4
#define SEQ 2048
#define NTOK (BATCH * SEQ)

typedef __attribute__((ext_vector_type(16))) __bf16 v16bf;
typedef __attribute__((ext_vector_type(8)))  float  v8f;
typedef uint32_t u32x4 __attribute__((ext_vector_type(4)));
typedef uint32_t u32x8 __attribute__((ext_vector_type(8)));

union Frag {
    v16bf v;
    uint32_t u[8];
};

__device__ __forceinline__ __bf16 f2bf(float f) {
    union { float f; uint32_t u; } x;
    x.f = f;
    uint32_t r = x.u + 0x7FFFu + ((x.u >> 16) & 1u);
    uint16_t h = (uint16_t)(r >> 16);
    __bf16 b;
    __builtin_memcpy(&b, &h, 2);
    return b;
}

// ---------------- pack kernels ----------------

__global__ void pack_bf16_kernel(const float* __restrict__ src,
                                 __bf16* __restrict__ dst, int n) {
    int i = blockIdx.x * blockDim.x + threadIdx.x;
    if (i < n) dst[i] = f2bf(src[i]);
}

// W row-major (K x N) f32  ->  col-major (N x K) bf16 (K contiguous per column)
__global__ void pack_w_colmajor_kernel(const float* __restrict__ w,
                                       __bf16* __restrict__ dst, int K, int N) {
    int i = blockIdx.x * blockDim.x + threadIdx.x;
    if (i >= K * N) return;
    int n = i % N;
    int k = i / N;
    dst[(size_t)n * K + k] = f2bf(w[i]);
}

// RoPE (interleaved pairs) + bf16 pack into (B, nheads, SEQ, HD)
__global__ void rope_pack_kernel(const float* __restrict__ src,   // (NTOK, nheads*HD)
                                 const float* __restrict__ cosb,  // (SEQ, HD/2)
                                 const float* __restrict__ sinb,
                                 __bf16* __restrict__ dst,        // (B, nheads, SEQ, HD)
                                 int nheads, float scale, int total) {
    int i = blockIdx.x * blockDim.x + threadIdx.x;
    if (i >= total) return;
    int p = i & 31;                 // pair index 0..31
    int h = (i >> 5) % nheads;
    int tok = i / (32 * nheads);
    int s = tok % SEQ;
    int b = tok / SEQ;
    float c = cosb[s * (HD / 2) + p];
    float sn = sinb[s * (HD / 2) + p];
    const float* sp = src + (size_t)tok * nheads * HD + h * HD + 2 * p;
    float x1 = sp[0], x2 = sp[1];
    float o1 = (x1 * c - x2 * sn) * scale;
    float o2 = (x1 * sn + x2 * c) * scale;
    __bf16* dp = dst + ((((size_t)b * nheads + h) * SEQ + s) * HD + 2 * p);
    dp[0] = f2bf(o1);
    dp[1] = f2bf(o2);
}

// V (NTOK, NKV*HD) f32 -> Vt (B, NKV, HD, SEQ) bf16  (key contiguous per dim row)
__global__ void pack_v_t_kernel(const float* __restrict__ vf,
                                __bf16* __restrict__ vt, int total) {
    int i = blockIdx.x * blockDim.x + threadIdx.x;
    if (i >= total) return;
    int c = i % (NKV * HD);
    int tok = i / (NKV * HD);
    int kv = c / HD;
    int d = c % HD;
    int s = tok % SEQ;
    int b = tok / SEQ;
    vt[(((size_t)b * NKV + kv) * HD + d) * SEQ + s] = f2bf(vf[i]);
}

// ---------------- bf16 WMMA GEMM ----------------
// C[M,N] = A[M,K] (bf16 row-major) @ B (bf16, stored as Bp[N][K] col-major) + bias
// block: 128 threads = 4 waves in 2x2; wave computes 32x32 via 2x2 WMMA tiles.
__global__ __launch_bounds__(128) void gemm_bf16_wmma_kernel(
    const __bf16* __restrict__ A,
    const __bf16* __restrict__ Bp,
    const float* __restrict__ bias,   // may be null
    float* __restrict__ C,
    int M, int N, int K) {
    const int lane = threadIdx.x & 31;
    const int wave = threadIdx.x >> 5;
    const int hf = lane >> 4;      // half of wave (0: lanes 0-15, 1: lanes 16-31)
    const int l16 = lane & 15;
    const int wr = wave >> 1, wc = wave & 1;
    const int M0 = blockIdx.y * 64 + wr * 32;
    const int N0 = blockIdx.x * 64 + wc * 32;

    const v8f zero8 = {0.f, 0.f, 0.f, 0.f, 0.f, 0.f, 0.f, 0.f};
    v8f acc[2][2];
    acc[0][0] = zero8; acc[0][1] = zero8; acc[1][0] = zero8; acc[1][1] = zero8;

    for (int k0 = 0; k0 < K; k0 += 32) {
        Frag a[2], b[2];
#pragma unroll
        for (int mt = 0; mt < 2; ++mt) {
            const __bf16* ab = A + (size_t)(M0 + mt * 16 + l16) * K + k0;
#pragma unroll
            for (int j = 0; j < 8; ++j) {
                // A operand 16x32, k order used here (same as B): K = e + 16*hf
                int kb = 2 * j + 16 * hf;
                a[mt].u[j] = *(const uint32_t*)(ab + kb);
            }
        }
#pragma unroll
        for (int nt = 0; nt < 2; ++nt) {
            const __bf16* bb = Bp + (size_t)(N0 + nt * 16 + l16) * K + k0 + 16 * hf;
#pragma unroll
            for (int j = 0; j < 8; ++j)  // B operand, k order used here (same as A): K = e + 16*hf, N = lane%16
                b[nt].u[j] = *(const uint32_t*)(bb + 2 * j);
        }
#pragma unroll
        for (int mt = 0; mt < 2; ++mt)
#pragma unroll
            for (int nt = 0; nt < 2; ++nt)
                acc[mt][nt] = __builtin_amdgcn_wmma_f32_16x16x32_bf16(
                    false, a[mt].v, false, b[nt].v, (short)0, acc[mt][nt],
                    false, false);
    }

#pragma unroll
    for (int mt = 0; mt < 2; ++mt)
#pragma unroll
        for (int nt = 0; nt < 2; ++nt)
#pragma unroll
            for (int r = 0; r < 8; ++r) {
                int row = M0 + mt * 16 + r + 8 * hf;   // C-layout: M = r + 8*hf
                int col = N0 + nt * 16 + l16;
                float v = acc[mt][nt][r];
                if (bias) v += bias[col];
                C[(size_t)row * N + col] = v;
            }
}

// ---------------- TDM descriptor helpers (CDNA5 ISA) ----

// group0: count=1, lds byte offset, 57-bit global byte address, type=2
__device__ __forceinline__ u32x4 tdm_g0(uint32_t lds_addr, uint64_t gaddr) {
    u32x4 g;
    g[0] = 1u;                                   // count=1, no gather
    g[1] = lds_addr;                             // [63:32] lds_addr
    g[2] = (uint32_t)gaddr;                      // [95:64] global_addr lo
    g[3] = ((uint32_t)(gaddr >> 32) & 0x01FFFFFFu) | (2u << 30);  // addr hi | type=2
    return g;
}

// group1: data_size=4B, 2D tile; dims/strides in dword units
__device__ __forceinline__ u32x8 tdm_g1(uint32_t tdim0, uint32_t tdim1,
                                        uint32_t tile0, uint32_t tile1,
                                        uint64_t stride0) {
    u32x8 g;
    g[0] = (2u << 16);                                    // wg_mask=0, data_size=4B
    g[1] = (tdim0 & 0xFFFFu) << 16;                       // [63:48] tensor_dim0 lo16
    g[2] = (tdim0 >> 16) | ((tdim1 & 0xFFFFu) << 16);     // dim0 hi16 | dim1 lo16
    g[3] = (tdim1 >> 16) | (tile0 << 16);                 // dim1 hi16 | tile_dim0
    g[4] = (tile1 & 0xFFFFu);                             // tile_dim1 (tile_dim2=0)
    g[5] = (uint32_t)stride0;                             // dim0_stride lo32
    g[6] = (uint32_t)((stride0 >> 32) & 0xFFFFu);         // dim0_stride hi16 (dim1_stride=0)
    g[7] = 0;
    return g;
}

__device__ __forceinline__ void tdm_load(u32x4 g0, u32x8 g1) {
    asm volatile("tensor_load_to_lds %0, %1" :: "s"(g0), "s"(g1) : "memory");
}

// ---------------- flash attention ----------------
// Block = 4 waves / 64 queries per (b,h). Wave 0 drives the Tensor Data Mover
// to stage a 64-key K chunk (contiguous 8KB) and a 64x64 V chunk (strided 2D)
// into LDS; all 4 waves compute WMMA fragments straight from LDS.
#define AKC 64   // keys per staged chunk

__global__ __launch_bounds__(128) void attn_wmma_kernel(
    const __bf16* __restrict__ Q,   // (B, NHEADS, SEQ, HD) pre-scaled by 1/8
    const __bf16* __restrict__ Kb,  // (B, NKV, SEQ, HD)
    const __bf16* __restrict__ Vt,  // (B, NKV, HD, SEQ)
    __bf16* __restrict__ O) {       // (NTOK, HIDDEN)
    extern __shared__ __align__(16) __bf16 sh[];
    // LDS layout (bf16 elements): [0,4096): K chunk 64x64; [4096,8192): V chunk 64x64;
    // [8192,+1024*wave): per-wave P tile 16x64
    __bf16* Kc = sh;
    __bf16* Vc = sh + 4096;

    const int lane = threadIdx.x & 31;
    const int wave = threadIdx.x >> 5;
    const int hf = lane >> 4;
    const int l16 = lane & 15;
    __bf16* Pw = sh + 8192 + wave * 1024;

    const int h = blockIdx.y;
    const int b = blockIdx.z;
    const int kv = h / NREP;
    const int qt = blockIdx.x * 4 + wave;    // 16-query tile index

    const __bf16* Qp = Q + (((size_t)b * NHEADS + h) * SEQ + qt * 16) * HD;
    const __bf16* Kp = Kb + ((size_t)b * NKV + kv) * SEQ * HD;
    const __bf16* Vp = Vt + ((size_t)b * NKV + kv) * HD * SEQ;

    // Load Q tile (16x64) once from global: two 16x32 A-fragments
    Frag q[2];
#pragma unroll
    for (int c = 0; c < 2; ++c) {
        const __bf16* ab = Qp + (size_t)l16 * HD + c * 32;
#pragma unroll
        for (int j = 0; j < 8; ++j) {
            int kb = 2 * j + 16 * hf;
            q[c].u[j] = *(const uint32_t*)(ab + kb);
        }
    }

    const v8f zero8 = {0.f, 0.f, 0.f, 0.f, 0.f, 0.f, 0.f, 0.f};
    v8f o[4];
    o[0] = zero8; o[1] = zero8; o[2] = zero8; o[3] = zero8;
    float mrow[8], lrow[8];
#pragma unroll
    for (int r = 0; r < 8; ++r) { mrow[r] = -1e30f; lrow[r] = 0.f; }

    const uint32_t lds_base = (uint32_t)(uintptr_t)(void*)sh;  // flat addr[31:0] = LDS offset
    const uint64_t kga = (uint64_t)(uintptr_t)Kp;
    const uint64_t vga = (uint64_t)(uintptr_t)Vp;

    for (int kc = 0; kc < SEQ; kc += AKC) {
        if (threadIdx.x < 32) {   // wave 0 drives the TDM
            // K chunk: 64 keys x 64 dims bf16 -> contiguous 8192B (2048 dwords)
            tdm_load(tdm_g0(lds_base, kga + (uint64_t)kc * (HD * 2)),
                     tdm_g1(/*tdim0=*/2048, /*tdim1=*/1, /*tile0=*/2048, /*tile1=*/1,
                            /*stride0=*/2048));
            // V chunk: 64 dim-rows x 64 keys (128B rows), row stride SEQ*2 bytes
            tdm_load(tdm_g0(lds_base + 8192u, vga + (uint64_t)kc * 2),
                     tdm_g1(/*tdim0=*/32, /*tdim1=*/HD, /*tile0=*/32, /*tile1=*/HD,
                            /*stride0=*/(SEQ * 2) / 4));
            __builtin_amdgcn_s_wait_tensorcnt(0);
        }
        __syncthreads();

        // scores: S(16x64) = Q(16x64) @ K^T(64x64), four 16x16 N-tiles from LDS
        v8f sc[4];
#pragma unroll
        for (int nt = 0; nt < 4; ++nt) {
            Frag kf0, kf1;   // B-operand: n = key, contraction = head dim
            const __bf16* kb_ = Kc + (size_t)(nt * 16 + l16) * HD + 16 * hf;
#pragma unroll
            for (int j = 0; j < 8; ++j) {
                kf0.u[j] = *(const uint32_t*)(kb_ + 2 * j);        // dims 0..31
                kf1.u[j] = *(const uint32_t*)(kb_ + 32 + 2 * j);   // dims 32..63
            }
            sc[nt] = zero8;
            sc[nt] = __builtin_amdgcn_wmma_f32_16x16x32_bf16(
                false, q[0].v, false, kf0.v, (short)0, sc[nt], false, false);
            sc[nt] = __builtin_amdgcn_wmma_f32_16x16x32_bf16(
                false, q[1].v, false, kf1.v, (short)0, sc[nt], false, false);
        }

        // online softmax over this 64-key chunk; rows r+8*hf live in halves
#pragma unroll
        for (int r = 0; r < 8; ++r) {
            float v0 = sc[0][r], v1 = sc[1][r], v2 = sc[2][r], v3 = sc[3][r];
            float mx = fmaxf(fmaxf(v0, v1), fmaxf(v2, v3));
#pragma unroll
            for (int m = 1; m < 16; m <<= 1) mx = fmaxf(mx, __shfl_xor(mx, m, 32));
            float mnew = fmaxf(mrow[r], mx);
            float rescale = __expf(mrow[r] - mnew);
            float p0 = __expf(v0 - mnew);
            float p1 = __expf(v1 - mnew);
            float p2 = __expf(v2 - mnew);
            float p3 = __expf(v3 - mnew);
            float rs = (p0 + p1) + (p2 + p3);
#pragma unroll
            for (int m = 1; m < 16; m <<= 1) rs += __shfl_xor(rs, m, 32);
            lrow[r] = lrow[r] * rescale + rs;
            mrow[r] = mnew;
#pragma unroll
            for (int t = 0; t < 4; ++t) o[t][r] *= rescale;
            int row = r + 8 * hf;
            Pw[row * 64 + l16]      = f2bf(p0);
            Pw[row * 64 + 16 + l16] = f2bf(p1);
            Pw[row * 64 + 32 + l16] = f2bf(p2);
            Pw[row * 64 + 48 + l16] = f2bf(p3);
        }

        // reload P(16x64) in A-fragment layout (per-wave region; LDS is in-order per wave)
        Frag pf[2];
        const __bf16* pb = Pw + (size_t)l16 * 64;
#pragma unroll
        for (int c = 0; c < 2; ++c)
#pragma unroll
            for (int j = 0; j < 8; ++j) {
                int kb = 2 * j + 16 * hf;
                pf[c].u[j] = *(const uint32_t*)(pb + c * 32 + kb);
            }

        // O(16x64) += P(16x64) @ V(64x64), four 16x16 N-tiles from LDS
#pragma unroll
        for (int t = 0; t < 4; ++t) {
            Frag vf0, vf1;   // B-operand: n = dim, contraction = key
            const __bf16* vb = Vc + (size_t)(t * 16 + l16) * 64 + 16 * hf;
#pragma unroll
            for (int j = 0; j < 8; ++j) {
                vf0.u[j] = *(const uint32_t*)(vb + 2 * j);        // keys 0..31
                vf1.u[j] = *(const uint32_t*)(vb + 32 + 2 * j);   // keys 32..63
            }
            o[t] = __builtin_amdgcn_wmma_f32_16x16x32_bf16(
                false, pf[0].v, false, vf0.v, (short)0, o[t], false, false);
            o[t] = __builtin_amdgcn_wmma_f32_16x16x32_bf16(
                false, pf[1].v, false, vf1.v, (short)0, o[t], false, false);
        }
        __syncthreads();   // all waves done with K/V chunk before next TDM overwrite
    }

    // normalize and write bf16 into (token, HIDDEN) for the output projection
#pragma unroll
    for (int r = 0; r < 8; ++r) {
        float inv = 1.0f / lrow[r];
        int srow = qt * 16 + r + 8 * hf;
        size_t tok = (size_t)b * SEQ + srow;
#pragma unroll
        for (int t = 0; t < 4; ++t) {
            float val = o[t][r] * inv;
            O[tok * HIDDEN + h * HD + t * 16 + l16] = f2bf(val);
        }
    }
}

// ---------------- launch ----------------

extern "C" void kernel_launch(void* const* d_in, const int* in_sizes, int n_in,
                              void* d_out, int out_size, void* d_ws, size_t ws_size,
                              hipStream_t stream) {
    const float* x    = (const float*)d_in[0];
    const float* fcos = (const float*)d_in[1];
    const float* fsin = (const float*)d_in[2];
    const float* Wq   = (const float*)d_in[3];
    const float* bq   = (const float*)d_in[4];
    const float* Wk   = (const float*)d_in[5];
    const float* bk   = (const float*)d_in[6];
    const float* Wv   = (const float*)d_in[7];
    const float* bv   = (const float*)d_in[8];
    const float* Wo   = (const float*)d_in[9];
    float* out = (float*)d_out;

    // workspace carve-up (256B aligned)
    size_t off = 0;
    auto carve = [&](size_t bytes) {
        void* p = (char*)d_ws + off;
        off += (bytes + 255) & ~(size_t)255;
        return p;
    };
    __bf16* Wqp = (__bf16*)carve((size_t)HIDDEN * HIDDEN * 2);
    __bf16* Wkp = (__bf16*)carve((size_t)HIDDEN * (NKV * HD) * 2);
    __bf16* Wvp = (__bf16*)carve((size_t)HIDDEN * (NKV * HD) * 2);
    __bf16* Wop = (__bf16*)carve((size_t)HIDDEN * HIDDEN * 2);
    __bf16* Xbf = (__bf16*)carve((size_t)NTOK * HIDDEN * 2);
    float*  Qf  = (float*)carve((size_t)NTOK * HIDDEN * 4);
    float*  Kf  = (float*)carve((size_t)NTOK * (NKV * HD) * 4);
    float*  Vf  = (float*)carve((size_t)NTOK * (NKV * HD) * 4);
    __bf16* Qbf = (__bf16*)carve((size_t)NTOK * HIDDEN * 2);
    __bf16* Kbf = (__bf16*)carve((size_t)NTOK * (NKV * HD) * 2);
    __bf16* Vtb = (__bf16*)carve((size_t)NTOK * (NKV * HD) * 2);
    __bf16* Abf = (__bf16*)carve((size_t)NTOK * HIDDEN * 2);
    (void)ws_size; (void)n_in; (void)in_sizes; (void)out_size;

    const int TPB = 256;

    // 1) pack activations + weights to bf16
    {
        int n = NTOK * HIDDEN;
        pack_bf16_kernel<<<(n + TPB - 1) / TPB, TPB, 0, stream>>>(x, Xbf, n);
        int nq = HIDDEN * HIDDEN;
        pack_w_colmajor_kernel<<<(nq + TPB - 1) / TPB, TPB, 0, stream>>>(Wq, Wqp, HIDDEN, HIDDEN);
        int nk = HIDDEN * (NKV * HD);
        pack_w_colmajor_kernel<<<(nk + TPB - 1) / TPB, TPB, 0, stream>>>(Wk, Wkp, HIDDEN, NKV * HD);
        pack_w_colmajor_kernel<<<(nk + TPB - 1) / TPB, TPB, 0, stream>>>(Wv, Wvp, HIDDEN, NKV * HD);
        pack_w_colmajor_kernel<<<(nq + TPB - 1) / TPB, TPB, 0, stream>>>(Wo, Wop, HIDDEN, HIDDEN);
    }

    // 2) QKV projections (bf16 WMMA, f32 out, fused bias)
    {
        dim3 blk(128);
        dim3 gq(HIDDEN / 64, NTOK / 64);
        gemm_bf16_wmma_kernel<<<gq, blk, 0, stream>>>(Xbf, Wqp, bq, Qf, NTOK, HIDDEN, HIDDEN);
        dim3 gk((NKV * HD) / 64, NTOK / 64);
        gemm_bf16_wmma_kernel<<<gk, blk, 0, stream>>>(Xbf, Wkp, bk, Kf, NTOK, NKV * HD, HIDDEN);
        gemm_bf16_wmma_kernel<<<gk, blk, 0, stream>>>(Xbf, Wvp, bv, Vf, NTOK, NKV * HD, HIDDEN);
    }

    // 3) RoPE + layout packs (Q pre-scaled by softmax scale 1/sqrt(64))
    {
        int tq = NTOK * NHEADS * (HD / 2);
        rope_pack_kernel<<<(tq + TPB - 1) / TPB, TPB, 0, stream>>>(
            Qf, fcos, fsin, Qbf, NHEADS, 0.125f, tq);
        int tk = NTOK * NKV * (HD / 2);
        rope_pack_kernel<<<(tk + TPB - 1) / TPB, TPB, 0, stream>>>(
            Kf, fcos, fsin, Kbf, NKV, 1.0f, tk);
        int tv = NTOK * NKV * HD;
        pack_v_t_kernel<<<(tv + TPB - 1) / TPB, TPB, 0, stream>>>(Vf, Vtb, tv);
    }

    // 4) attention: 4 waves/block, TDM-staged K/V chunks, 24KB dynamic LDS
    {
        dim3 grid(SEQ / AKC, NHEADS, BATCH);
        attn_wmma_kernel<<<grid, dim3(128), 24576, stream>>>(Qbf, Kbf, Vtb, Abf);
    }

    // 5) output projection straight to d_out (no bias)
    {
        dim3 blk(128);
        dim3 go(HIDDEN / 64, NTOK / 64);
        gemm_bf16_wmma_kernel<<<go, blk, 0, stream>>>(Abf, Wop, nullptr, out, NTOK, HIDDEN, HIDDEN);
    }
}
